// BertSelfAttention_2551210573998
// MI455X (gfx1250) — matrix operand order repaired and checked by reading, not compiled, not run on hardware
//
#include <hip/hip_runtime.h>

typedef __attribute__((ext_vector_type(16))) _Float16 v16h;
typedef __attribute__((ext_vector_type(8)))  _Float16 v8h;
typedef __attribute__((ext_vector_type(4)))  _Float16 v4h;
typedef __attribute__((ext_vector_type(8)))  float    v8f;
typedef __attribute__((ext_vector_type(4)))  float    v4f;

#define WMMA_F16(a, b, c) \
  __builtin_amdgcn_wmma_f32_16x16x32_f16(false, (a), false, (b), (short)0, (c), false, false)

constexpr int BB = 2, S = 2048, D = 768, H = 12, DH = 64;
constexpr int BS = BB * S;                               // 4096
constexpr size_t HS16_OFF = 0;                           // hs as f16: 4096*768
constexpr size_t WT_SZ   = (size_t)D * D;                // 589824
constexpr size_t WT_OFF  = (size_t)BS * D;               // 3145728
constexpr size_t QKV_SZ  = (size_t)BB * H * S * DH;      // 3145728
constexpr size_t Q_OFF   = WT_OFF + 3 * WT_SZ;           // 4915200
constexpr size_t K_OFF   = Q_OFF + QKV_SZ;               // 8060928
constexpr size_t VT_OFF  = K_OFF + QKV_SZ;               // 11206656
constexpr size_t CTX_ELEMS = (size_t)BS * D;             // context precedes softmax in d_out

constexpr int LPITCH = 72;                               // padded LDS row (halfs)
constexpr int LBUF   = 64 * LPITCH;                      // one ping-pong buffer (halfs)

__device__ __forceinline__ unsigned int pack_h2(float x, float y) {
  _Float16 hx = (_Float16)x, hy = (_Float16)y;
  unsigned short ux = __builtin_bit_cast(unsigned short, hx);
  unsigned short uy = __builtin_bit_cast(unsigned short, hy);
  return (unsigned int)ux | ((unsigned int)uy << 16);
}

// ---------------- prep: hidden_states f32 -> f16 -------------------------
__global__ __launch_bounds__(256) void k_cvt_hs(const float* __restrict__ hs,
                                                _Float16* __restrict__ o) {
  int i = blockIdx.x * 256 + threadIdx.x;          // exactly BS*D/4 threads
  v4f x = ((const v4f*)hs)[i];
  v4h y;
  y[0] = (_Float16)x[0]; y[1] = (_Float16)x[1];
  y[2] = (_Float16)x[2]; y[3] = (_Float16)x[3];
  ((v4h*)o)[i] = y;
}

// ---------------- prep: W [in][out] f32 -> W^T [out][in] f16 -------------
__global__ __launch_bounds__(256) void k_wt(const float* __restrict__ Wq,
                                            const float* __restrict__ Wk,
                                            const float* __restrict__ Wv,
                                            _Float16* __restrict__ wt) {
  int gid = blockIdx.x * 256 + threadIdx.x;        // exactly D*D threads per z
  int n = gid / D, k = gid - n * D;
  const float* W = (blockIdx.z == 0) ? Wq : ((blockIdx.z == 1) ? Wk : Wv);
  wt[(size_t)blockIdx.z * WT_SZ + gid] = (_Float16)W[k * D + n];
}

// ---------------- fused QKV projection GEMM (WMMA f16, f32 acc) ----------
// grid (32, 12, 3), block 256 = 8 waves; wave tile: 16 rows x 64 cols.
// Ping-pong register pipeline (unroll-by-2), no cross-buffer copies.
__global__ __launch_bounds__(256) void k_qkv(_Float16* ws,
                                             const float* __restrict__ bq,
                                             const float* __restrict__ bk,
                                             const float* __restrict__ bv) {
  const int tid = threadIdx.x, wave = tid >> 5, lane = tid & 31;
  const int lo = lane & 15, hi = lane >> 4;
  const int mat = blockIdx.z;
  const int mbase = blockIdx.x * 128 + wave * 16;
  const int nbase = blockIdx.y * 64;

  const _Float16* A  = ws + HS16_OFF + (size_t)(mbase + lo) * D;
  const _Float16* Wt = ws + WT_OFF + (size_t)mat * WT_SZ;
  const float* bias = (mat == 0) ? bq : ((mat == 1) ? bk : bv);

  auto loadA = [&](int kc) -> v16h {
    v8h a0 = *(const v8h*)(A + kc + hi * 16);
    v8h a1 = *(const v8h*)(A + kc + hi * 16 + 8);
    v16h av;
#pragma unroll
    for (int i = 0; i < 8; ++i) { av[i] = a0[i]; av[8 + i] = a1[i]; }
    return av;
  };
  auto loadB = [&](int kc, int nt) -> v16h {
    return *(const v16h*)(Wt + (size_t)(nbase + nt * 16 + lo) * D + kc + hi * 16);
  };

  v8f acc[4] = {};
  v16h a0 = loadA(0), a1;
  v16h b0[4], b1[4];
#pragma unroll
  for (int nt = 0; nt < 4; ++nt) b0[nt] = loadB(0, nt);

  for (int kc = 32; kc < D; kc += 64) {           // MMAs steps kc-32 and kc; loads kc and kc+32
    a1 = loadA(kc);
#pragma unroll
    for (int nt = 0; nt < 4; ++nt) b1[nt] = loadB(kc, nt);
#pragma unroll
    for (int nt = 0; nt < 4; ++nt) acc[nt] = WMMA_F16(a0, b0[nt], acc[nt]);
    if (kc + 32 < D) {
      a0 = loadA(kc + 32);
#pragma unroll
      for (int nt = 0; nt < 4; ++nt) b0[nt] = loadB(kc + 32, nt);
    }
#pragma unroll
    for (int nt = 0; nt < 4; ++nt) acc[nt] = WMMA_F16(a1, b1[nt], acc[nt]);
  }

#pragma unroll
  for (int nt = 0; nt < 4; ++nt) {
    const int n = nbase + nt * 16 + lo;
    const float bsv = bias[n];
    const int hh = n >> 6, dh = n & 63;
    const int gm0 = mbase + hi * 8;               // rows gm0..gm0+7, same b (16 | 2048)
    const int bb = gm0 >> 11, ssp0 = gm0 & 2047;
    if (mat == 2) {
      // V^T rows are contiguous along s: one packed 16B store
      v8h pk;
#pragma unroll
      for (int r = 0; r < 8; ++r) pk[r] = (_Float16)(acc[nt][r] + bsv);
      *(v8h*)(ws + VT_OFF + ((size_t)((bb * H + hh) * DH + dh) << 11) + ssp0) = pk;
    } else {
      _Float16* dst = ws + (mat == 0 ? Q_OFF : K_OFF) +
                      ((size_t)((bb * H + hh) * S + ssp0) << 6) + dh;
#pragma unroll
      for (int r = 0; r < 8; ++r) dst[(size_t)r << 6] = (_Float16)(acc[nt][r] + bsv);
    }
  }
}

// ---------------- attention: S^T = K*Q^T, online softmax, C^T = V^T*P^T ---
// grid 384, block 256 = 8 waves; all 8 waves share (b,h). K / V^T chunks
// (64 keys x 64 dh f16) staged through double-buffered LDS; the global loads
// for chunk n+1 issue under chunk n's WMMAs.
__global__ __launch_bounds__(256) void k_attn(const _Float16* __restrict__ ws,
                                              const float* __restrict__ mask,
                                              float* __restrict__ out) {
  __shared__ _Float16 Kl[2][LBUF];
  __shared__ _Float16 Vl[2][LBUF];

  const int tid = threadIdx.x, wave = tid >> 5, lane = tid & 31;
  const int lo = lane & 15, hi = lane >> 4;
  const int w = blockIdx.x * 8 + wave;            // 0..3071
  const int b = w / (H * (S / 16));
  const int r0 = w - b * (H * (S / 16));
  const int h = r0 >> 7;
  const int qbase = (r0 & 127) << 4;

  const size_t bh = (size_t)(b * H + h);          // uniform across the block
  const _Float16* Qb = ws + Q_OFF + ((bh * S) << 6);
  const _Float16* Kb = ws + K_OFF + ((bh * S) << 6);
  const _Float16* Vb = ws + VT_OFF + ((bh * DH) << 11);
  const float* mrow = mask + b * S;
  float* osm = out + CTX_ELEMS + ((bh * S + qbase) << 11);

  // cooperative staging map: 4 threads per row, 16 halfs (32B) per thread
  const int srow = tid >> 2;                      // 0..63
  const int scol = (tid & 3) << 4;                // 0,16,32,48
  const _Float16* kg = Kb + (size_t)srow * DH + scol;   // + kc*DH per chunk
  const _Float16* vg = Vb + (size_t)srow * S + scol;    // + kc per chunk
  _Float16* kls = &Kl[0][srow * LPITCH + scol];
  _Float16* vls = &Vl[0][srow * LPITCH + scol];

  // Q^T B-operand: lane lo holds column q; halves j -> dh = c + hi*16 + j
  const _Float16* qr = Qb + (size_t)(qbase + lo) * DH;
  const v16h q0 = *(const v16h*)(qr + hi * 16);
  const v16h q1 = *(const v16h*)(qr + 32 + hi * 16);

  // one 16-key score tile from LDS buffer KB
  auto score = [&](const _Float16* KB, int kc, int klb, float* s) {
    const _Float16* kr = KB + (klb + lo) * LPITCH;
    v8h x0 = *(const v8h*)(kr + hi * 16);
    v8h x1 = *(const v8h*)(kr + hi * 16 + 8);
    v8h x2 = *(const v8h*)(kr + 32 + hi * 16);
    v8h x3 = *(const v8h*)(kr + 40 + hi * 16);
    v16h a0, a1;
#pragma unroll
    for (int i = 0; i < 8; ++i) { a0[i] = x0[i]; a0[8 + i] = x1[i]; a1[i] = x2[i]; a1[8 + i] = x3[i]; }
    v8f c = {};
    c = WMMA_F16(a0, q0, c);
    c = WMMA_F16(a1, q1, c);
    v4f m0 = *(const v4f*)(mrow + kc + klb + hi * 8);
    v4f m1 = *(const v4f*)(mrow + kc + klb + hi * 8 + 4);
#pragma unroll
    for (int r = 0; r < 4; ++r) {
      s[r]     = c[r]     * 0.125f + m0[r];       // 1/sqrt(64) + additive mask
      s[4 + r] = c[4 + r] * 0.125f + m1[r];
    }
  };

  // ---- pass A: row max + rescaled row sum ------------------------------
  float mmax = -3.0e38f, lsum = 0.f;
  {
    v8h ka0 = *(const v8h*)(kg);
    v8h ka1 = *(const v8h*)(kg + 8);
    int cur = 0;
    for (int kc = 0; kc < S; kc += 64) {
      __syncthreads();
      *(v8h*)(kls + cur * LBUF)     = ka0;
      *(v8h*)(kls + cur * LBUF + 8) = ka1;
      if (kc + 64 < S) {
        ka0 = *(const v8h*)(kg + (size_t)(kc + 64) * DH);
        ka1 = *(const v8h*)(kg + (size_t)(kc + 64) * DH + 8);
      }
      if (kc + 128 < S) __builtin_prefetch(kg + (size_t)(kc + 128) * DH, 0, 3);
      __syncthreads();
      const _Float16* KB = &Kl[cur][0];
#pragma unroll
      for (int t4 = 0; t4 < 4; ++t4) {
        float s[8];
        score(KB, kc, t4 * 16, s);
        float tm = s[0];
#pragma unroll
        for (int r = 1; r < 8; ++r) tm = fmaxf(tm, s[r]);
        tm = fmaxf(tm, __shfl_xor(tm, 16, 32));
        const float mn = fmaxf(mmax, tm);
        const float corr = __expf(mmax - mn);
        float ps = 0.f;
#pragma unroll
        for (int r = 0; r < 8; ++r) ps += __expf(s[r] - mn);
        lsum = lsum * corr + ps;
        mmax = mn;
      }
      cur ^= 1;
    }
  }
  lsum += __shfl_xor(lsum, 16, 32);
  const float inv_l = 1.0f / lsum;

  // ---- pass B: normalized P -> d_out softmax, and C^T = V^T * P^T ------
  v8f acc[4] = {};
  {
    v8h ka0 = *(const v8h*)(kg);
    v8h ka1 = *(const v8h*)(kg + 8);
    v8h va0 = *(const v8h*)(vg);
    v8h va1 = *(const v8h*)(vg + 8);
    int cur = 0;
    for (int kc = 0; kc < S; kc += 64) {
      __syncthreads();
      *(v8h*)(kls + cur * LBUF)     = ka0;
      *(v8h*)(kls + cur * LBUF + 8) = ka1;
      *(v8h*)(vls + cur * LBUF)     = va0;
      *(v8h*)(vls + cur * LBUF + 8) = va1;
      if (kc + 64 < S) {
        ka0 = *(const v8h*)(kg + (size_t)(kc + 64) * DH);
        ka1 = *(const v8h*)(kg + (size_t)(kc + 64) * DH + 8);
        va0 = *(const v8h*)(vg + kc + 64);
        va1 = *(const v8h*)(vg + kc + 64 + 8);
      }
      if (kc + 128 < S) {
        __builtin_prefetch(kg + (size_t)(kc + 128) * DH, 0, 3);
        __builtin_prefetch(vg + kc + 128, 0, 3);
      }
      __syncthreads();
      const _Float16* KB = &Kl[cur][0];
      const _Float16* VB = &Vl[cur][0];
#pragma unroll
      for (int half = 0; half < 2; ++half) {
        const int klb = half * 32;                // 32-key group within chunk
        float s0[8], s1[8];
        score(KB, kc, klb, s0);
        score(KB, kc, klb + 16, s1);
        float p0[8], p1[8];
#pragma unroll
        for (int r = 0; r < 8; ++r) {
          p0[r] = __expf(s0[r] - mmax) * inv_l;
          p1[r] = __expf(s1[r] - mmax) * inv_l;
        }
        // softmax output: lane writes 8+8 consecutive keys of row q = qbase+lo
        float* orow = osm + (size_t)lo * S + kc + klb + hi * 8;
        v4f w0 = {p0[0], p0[1], p0[2], p0[3]};
        v4f w1 = {p0[4], p0[5], p0[6], p0[7]};
        v4f w2 = {p1[0], p1[1], p1[2], p1[3]};
        v4f w3 = {p1[4], p1[5], p1[6], p1[7]};
        *(v4f*)(orow)      = w0;
        *(v4f*)(orow + 4)  = w1;
        *(v4f*)(orow + 16) = w2;
        *(v4f*)(orow + 20) = w3;

        // Re-pack P into B-operand (32 x 16): lane lo = q, K = keys.
        unsigned int a0 = pack_h2(p0[0], p0[1]), a1 = pack_h2(p0[2], p0[3]);
        unsigned int a2 = pack_h2(p0[4], p0[5]), a3 = pack_h2(p0[6], p0[7]);
        unsigned int b0 = pack_h2(p1[0], p1[1]), b1 = pack_h2(p1[2], p1[3]);
        unsigned int b2 = pack_h2(p1[4], p1[5]), b3 = pack_h2(p1[6], p1[7]);
        unsigned int t0 = (unsigned int)__shfl_xor((int)(hi ? a0 : b0), 16, 32);
        unsigned int t1 = (unsigned int)__shfl_xor((int)(hi ? a1 : b1), 16, 32);
        unsigned int t2 = (unsigned int)__shfl_xor((int)(hi ? a2 : b2), 16, 32);
        unsigned int t3 = (unsigned int)__shfl_xor((int)(hi ? a3 : b3), 16, 32);
        union { v16h hv; unsigned int u[8]; } pb;
        pb.u[0] = hi ? t0 : a0; pb.u[1] = hi ? t1 : a1;
        pb.u[2] = hi ? t2 : a2; pb.u[3] = hi ? t3 : a3;
        pb.u[4] = hi ? b0 : t0; pb.u[5] = hi ? b1 : t1;
        pb.u[6] = hi ? b2 : t2; pb.u[7] = hi ? b3 : t3;

#pragma unroll
        for (int t = 0; t < 4; ++t) {
          const _Float16* vr = VB + (t * 16 + lo) * LPITCH + klb;
          v8h y0 = *(const v8h*)(vr + hi * 16);
          v8h y1 = *(const v8h*)(vr + hi * 16 + 8);
          v16h va;
#pragma unroll
          for (int i = 0; i < 8; ++i) { va[i] = y0[i]; va[8 + i] = y1[i]; }
          acc[t] = WMMA_F16(va, pb.hv, acc[t]);
        }
      }
      cur ^= 1;
    }
  }

  // context store: C^T D-layout lane = q, vgpr r = dh offset -> contiguous floats
  float* oc = out + (size_t)(b * S + qbase + lo) * D + h * DH + hi * 8;
#pragma unroll
  for (int t = 0; t < 4; ++t) {
    v4f c0 = {acc[t][0], acc[t][1], acc[t][2], acc[t][3]};
    v4f c1 = {acc[t][4], acc[t][5], acc[t][6], acc[t][7]};
    *(v4f*)(oc + t * 16)     = c0;
    *(v4f*)(oc + t * 16 + 4) = c1;
  }
}

extern "C" void kernel_launch(void* const* d_in, const int* in_sizes, int n_in,
                              void* d_out, int out_size, void* d_ws, size_t ws_size,
                              hipStream_t stream) {
  (void)in_sizes; (void)n_in; (void)out_size; (void)ws_size;
  const float* hs   = (const float*)d_in[0];
  const float* mask = (const float*)d_in[1];
  const float* Wq   = (const float*)d_in[2];
  const float* bq   = (const float*)d_in[3];
  const float* Wk   = (const float*)d_in[4];
  const float* bk   = (const float*)d_in[5];
  const float* Wv   = (const float*)d_in[6];
  const float* bv   = (const float*)d_in[7];
  _Float16* ws = (_Float16*)d_ws;
  float* out = (float*)d_out;

  k_cvt_hs<<<(BS * D) / (4 * 256), 256, 0, stream>>>(hs, ws + HS16_OFF);
  k_wt<<<dim3((D * D) / 256, 1, 3), 256, 0, stream>>>(Wq, Wk, Wv, ws + WT_OFF);
  k_qkv<<<dim3(BS / 128, D / 64, 3), 256, 0, stream>>>(ws, bq, bk, bv);
  k_attn<<<(BB * H * (S / 16)) / 8, 256, 0, stream>>>(ws, mask, out);
}
